// TinyMod_69380901700330
// MI455X (gfx1250) — matrix operand order repaired and checked by reading, not compiled, not run on hardware
//
#include <hip/hip_runtime.h>

// ---------------------------------------------------------------------------
// TinyMod fused kernel for MI455X (gfx1250, wave32, WMMA).
//
// Roofline: 12.9 GFLOP vs 270MB mandatory HBM traffic (x in, out out) ->
// memory bound (~12us @ 23.3TB/s) only if matrix math exceeds ~1.1 PFLOP/s,
// which requires the bf16 16x16x32 WMMA path (f32 accumulate). FP32 WMMA
// (16x16x4) is ~8x lower throughput per instruction and would dominate.
//
// Round-2 change: weights are pre-converted to bf16 in d_ws by a prep kernel
// so the hot stage-4 loop does NO f32->bf16 VALU work (B frags are two
// global_load_b128 of L2-resident bf16). D-fragment stores use bases with
// the lane-dependent row folded in so offsets become immediates.
// ---------------------------------------------------------------------------

typedef __attribute__((ext_vector_type(16))) __bf16 v16bf;
typedef __attribute__((ext_vector_type(8)))  float  v8f;

union Frag {
  v16bf v;
  uint4 q[2];
};

// Problem constants (static per reference).
#define BSZ    8192
#define IDIM   4096
#define MSIZE  128
#define MCOUNT 32
#define BT     16          // batch rows per workgroup (one WMMA M tile)

// LDS pitches (bytes). Rows 16B-aligned; pitches de-phase banks.
#define XS_ROW   80        // 32 bf16 (64B) + 16B pad : xs[m][b][i], y2s[m][b][c]
#define XS_SLAB  (BT * XS_ROW)        // 1280
#define XS_SIZE  (MSIZE * XS_SLAB)    // 163840
#define Y1_ROW   272       // 128 bf16 (256B) + 16B pad : y1s[c][b][m]
#define Y1_SLAB  (BT * Y1_ROW)        // 4352
#define Y1_SIZE  (MCOUNT * Y1_SLAB)   // 139264
#define SMEM_BYTES (XS_SIZE + Y1_SIZE) // 303104 (<320KB/WGP)

// Workspace layout (bf16 elements).
#define WMODS_ELEMS (MCOUNT * MSIZE * MSIZE)   // 524288
#define WIN_OFF     (WMODS_ELEMS)              // 1024 elems
#define WOUT_OFF    (WMODS_ELEMS + 1024)       // 1024 elems
#define WS_ELEMS    (WMODS_ELEMS + 2048)       // 526336 (* 2B ~ 1.03MB)

// ---- Prep: convert W_mods / W_in / W_out fp32 -> bf16 into workspace -------
__global__ __launch_bounds__(256) void prep_bf16(
    const float* __restrict__ Wmods, const float* __restrict__ Win,
    const float* __restrict__ Wout, __bf16* __restrict__ wsb) {
  int idx = blockIdx.x * 256 + threadIdx.x;      // one thread per 8 elements
  const int n8 = WS_ELEMS / 8;
  if (idx >= n8) return;
  int e = idx * 8;
  const float* src;
  if (e < WMODS_ELEMS)            src = Wmods + e;
  else if (e < WMODS_ELEMS + 1024) src = Win  + (e - WMODS_ELEMS);
  else                             src = Wout + (e - WMODS_ELEMS - 1024);
  float4 a = ((const float4*)src)[0];
  float4 b = ((const float4*)src)[1];
  union { __bf16 h[8]; uint4 q; } o;
  o.h[0] = (__bf16)a.x; o.h[1] = (__bf16)a.y; o.h[2] = (__bf16)a.z; o.h[3] = (__bf16)a.w;
  o.h[4] = (__bf16)b.x; o.h[5] = (__bf16)b.y; o.h[6] = (__bf16)b.z; o.h[7] = (__bf16)b.w;
  ((uint4*)wsb)[idx] = o.q;
}

// ---- Main fused kernel ------------------------------------------------------
__global__ __launch_bounds__(256) void tinymod_kernel(
    const float* __restrict__ x,     const float* __restrict__ bin,
    const float* __restrict__ bmods, const float* __restrict__ bout,
    const __bf16* __restrict__ wsb,  float* __restrict__ out) {
  __shared__ __align__(16) unsigned char smem[SMEM_BYTES];
  unsigned char* xs  = smem;            // stage2 A; reused as y2s for stage6 A
  unsigned char* y1s = smem + XS_SIZE;  // stage4 A

  const int tid   = threadIdx.x;
  const int lane  = tid & 31;
  const int wave  = tid >> 5;           // 8 waves
  const int row16 = lane & 15;          // fragment row / N-column selector
  const int selA  = (lane & 16);        // A-frag byte off: lanes>=16 hold k{8..15,24..31}
  const int koffB = (lane >> 4) * 16;   // B-frag byte off: lanes>=16 hold k{8..15,24..31}
  const int brow  = (lane >> 4) * 8;    // C/D frag: lanes>=16 hold M = vgpr+8
  const int b0    = blockIdx.x * BT;

  const unsigned char* wmb  = (const unsigned char*)wsb;              // W_mods bf16
  const unsigned char* winb = (const unsigned char*)(wsb + WIN_OFF);  // W_in bf16
  const unsigned char* woub = (const unsigned char*)(wsb + WOUT_OFF); // W_out bf16

  // ---- Stage 1: x tile [16 x 4096] fp32 -> xs[m][b][i] bf16 (transpose).
  {
    int b   = tid >> 4;                  // each thread owns one batch row
    int col = tid & 15;
    const float4* xrow = (const float4*)(x + (size_t)(b0 + b) * IDIM);
    unsigned char* xb = xs + b * XS_ROW;
    for (int t = 0; t < 64; ++t) {
      int j4 = col + t * 16;             // float4 index in row (coalesced/halfwave)
      float4 v = xrow[j4];
      int i = j4 >> 5;                   // block index (k of stage 2)
      int m = (j4 & 31) << 2;            // within-block pos (m..m+3)
      unsigned char* p = xb + (size_t)m * XS_SLAB + i * 2;
      *(__bf16*)(p)               = (__bf16)v.x;
      *(__bf16*)(p + XS_SLAB)     = (__bf16)v.y;   // imm offsets 1280/2560/3840
      *(__bf16*)(p + 2 * XS_SLAB) = (__bf16)v.z;
      *(__bf16*)(p + 3 * XS_SLAB) = (__bf16)v.w;
    }
  }
  __syncthreads();

  // ---- Stage 2: y1[b,m,c] = x_blk[b,m,:] . W_in[c,:] + b_in[c]
  Frag  winf[2];
  float bias_in[2];
  for (int ct = 0; ct < 2; ++ct) {
    int n = ct * 16 + row16;
    const uint4* p = (const uint4*)(winb + n * 64 + koffB);
    winf[ct].q[0] = p[0];
    winf[ct].q[1] = p[2];
    bias_in[ct] = bin[n];
  }
  for (int r = 0; r < 16; ++r) {
    int m = wave * 16 + r;
    const unsigned char* arow = xs + (size_t)m * XS_SLAB + row16 * XS_ROW + selA;
    Frag a;
    a.q[0] = *(const uint4*)(arow);
    a.q[1] = *(const uint4*)(arow + 32);
    for (int ct = 0; ct < 2; ++ct) {
      v8f acc;
#pragma unroll
      for (int k = 0; k < 8; ++k) acc[k] = bias_in[ct];
      acc = __builtin_amdgcn_wmma_f32_16x16x32_bf16(false, a.v, false, winf[ct].v,
                                                    (short)0, acc, false, false);
      int c = ct * 16 + row16;
      unsigned char* dst = y1s + (size_t)c * Y1_SLAB + brow * Y1_ROW + (size_t)m * 2;
#pragma unroll
      for (int k = 0; k < 8; ++k)        // D[M=brow+k, N=c] -> y1s[c][b][m], imm offs
        *(__bf16*)(dst + k * Y1_ROW) = (__bf16)acc[k];
    }
  }
  __syncthreads();

  // ---- Stage 4 (dominant): per c: Y2_c[b,n] = y1s[c][b,:].W_mods[c,n,:] + b_mods
  for (int cc = 0; cc < 4; ++cc) {
    int c = wave * 4 + cc;
    const unsigned char* abase = y1s + (size_t)c * Y1_SLAB + row16 * Y1_ROW + selA;
    Frag af[4];
#pragma unroll
    for (int kk = 0; kk < 4; ++kk) {
      af[kk].q[0] = *(const uint4*)(abase + kk * 64);
      af[kk].q[1] = *(const uint4*)(abase + kk * 64 + 32);
    }
    for (int nt = 0; nt < 8; ++nt) {
      int n = nt * 16 + row16;
      const unsigned char* bbase = wmb + ((size_t)(c * MSIZE + n) * MSIZE) * 2 + koffB;
      Frag bf[4];
#pragma unroll
      for (int kk = 0; kk < 4; ++kk) {   // bf16 B rows: two b128 loads each, imm offs
        bf[kk].q[0] = *(const uint4*)(bbase + kk * 64);
        bf[kk].q[1] = *(const uint4*)(bbase + kk * 64 + 32);
      }
      float bias = bmods[c * MSIZE + n];
      v8f acc;
#pragma unroll
      for (int k = 0; k < 8; ++k) acc[k] = bias;
#pragma unroll
      for (int kk = 0; kk < 4; ++kk)
        acc = __builtin_amdgcn_wmma_f32_16x16x32_bf16(false, af[kk].v, false, bf[kk].v,
                                                      (short)0, acc, false, false);
      unsigned char* dst = xs + (size_t)n * XS_SLAB + brow * XS_ROW + (size_t)c * 2;
#pragma unroll
      for (int k = 0; k < 8; ++k)        // D -> y2s[n][b][c], imm offs k*80
        *(__bf16*)(dst + k * XS_ROW) = (__bf16)acc[k];
    }
  }
  __syncthreads();

  // ---- Stage 6: per m: out[b, m*32+o] = y2s[m][b,:] . W_out[o,:] + b_out[o]
  Frag  woutf[2];
  float bias_out[2];
  for (int ct = 0; ct < 2; ++ct) {
    int o = ct * 16 + row16;
    const uint4* p = (const uint4*)(woub + o * 64 + koffB);
    woutf[ct].q[0] = p[0];
    woutf[ct].q[1] = p[2];
    bias_out[ct] = bout[o];
  }
  for (int r = 0; r < 16; ++r) {
    int m = wave * 16 + r;
    const unsigned char* arow = xs + (size_t)m * XS_SLAB + row16 * XS_ROW + selA;
    Frag a;
    a.q[0] = *(const uint4*)(arow);
    a.q[1] = *(const uint4*)(arow + 32);
    for (int ct = 0; ct < 2; ++ct) {
      v8f acc;
#pragma unroll
      for (int k = 0; k < 8; ++k) acc[k] = bias_out[ct];
      acc = __builtin_amdgcn_wmma_f32_16x16x32_bf16(false, a.v, false, woutf[ct].v,
                                                    (short)0, acc, false, false);
      int o = ct * 16 + row16;
      float* orow = out + (size_t)(b0 + brow) * IDIM + (size_t)m * 32 + o;
#pragma unroll
      for (int k = 0; k < 8; ++k)        // imm offsets k*16384 bytes
        orow[(size_t)k * IDIM] = acc[k];
    }
  }
}

extern "C" void kernel_launch(void* const* d_in, const int* in_sizes, int n_in,
                              void* d_out, int out_size, void* d_ws, size_t ws_size,
                              hipStream_t stream) {
  (void)in_sizes; (void)n_in; (void)out_size; (void)ws_size;
  const float* x     = (const float*)d_in[0];
  const float* Win   = (const float*)d_in[1];
  const float* bin   = (const float*)d_in[2];
  const float* Wmods = (const float*)d_in[3];
  const float* bmods = (const float*)d_in[4];
  const float* Wout  = (const float*)d_in[5];
  const float* bout  = (const float*)d_in[6];
  float* out   = (float*)d_out;
  __bf16* wsb  = (__bf16*)d_ws;   // needs ~1.03MB of workspace

  prep_bf16<<<dim3((WS_ELEMS / 8 + 255) / 256), dim3(256), 0, stream>>>(
      Wmods, Win, Wout, wsb);
  tinymod_kernel<<<dim3(BSZ / BT), dim3(256), 0, stream>>>(
      x, bin, bmods, bout, wsb, out);
}
